// GNNAggregationForNegativeModule_4947802325002
// MI455X (gfx1250) — matrix operand order repaired and checked by reading, not compiled, not run on hardware
//
#include <hip/hip_runtime.h>
#include <hip/hip_bf16.h>

// Problem constants (match reference)
#define NN 8192   // nodes
#define DD 256    // feature dim
#define BB 4096   // batch
#define LL 3      // MLP layers

typedef __bf16 bf16;
typedef __attribute__((ext_vector_type(16))) __bf16 v16bf;
typedef __attribute__((ext_vector_type(8)))  float  v8f;
typedef __attribute__((ext_vector_type(4)))  float  f4;

// ---------------------------------------------------------------------------
// A-fragment (contiguous k order, same as B): elements 0..15 = K kbase..kbase+15,
// kbase = k + 16*(lane>>4).
// Loads 16 f32, converts to bf16 (v_cvt_pk_bf16_f32 pairs).
// ---------------------------------------------------------------------------
__device__ __forceinline__ v16bf load_cvt_afrag(const float* __restrict__ p) {
    f4 f0 = *(const f4*)(p);
    f4 f1 = *(const f4*)(p + 4);
    f4 f2 = *(const f4*)(p + 8);
    f4 f3 = *(const f4*)(p + 12);
    v16bf a;
    a[0]  = (bf16)f0[0]; a[1]  = (bf16)f0[1]; a[2]  = (bf16)f0[2]; a[3]  = (bf16)f0[3];
    a[4]  = (bf16)f1[0]; a[5]  = (bf16)f1[1]; a[6]  = (bf16)f1[2]; a[7]  = (bf16)f1[3];
    a[8]  = (bf16)f2[0]; a[9]  = (bf16)f2[1]; a[10] = (bf16)f2[2]; a[11] = (bf16)f2[3];
    a[12] = (bf16)f3[0]; a[13] = (bf16)f3[1]; a[14] = (bf16)f3[2]; a[15] = (bf16)f3[3];
    return a;
}

// B-fragment set: 8 tiles, each 32 contiguous bytes at bbase + t*16*NN + k.
__device__ __forceinline__ void load_bset(const bf16* __restrict__ bbase, int k,
                                          v16bf* __restrict__ dst) {
    #pragma unroll
    for (int t = 0; t < 8; ++t)
        dst[t] = *(const v16bf*)(bbase + (size_t)t * (16 * NN) + k);
}

__device__ __forceinline__ v8f wmma_bf16(v16bf a, v16bf b, v8f c) {
    return __builtin_amdgcn_wmma_f32_16x16x32_bf16(false, a, false, b,
                                                   (short)0, c, false, false);
}

// ---------------------------------------------------------------------------
// 1) Stable partition: positives first, preserving order (serial, 4096 iters).
//    Emits labels_sorted as float into d_out[0..B).
// ---------------------------------------------------------------------------
__global__ void sort_kernel(const int* __restrict__ labels,
                            const int* __restrict__ nidx,
                            int* __restrict__ idx_s,
                            int* __restrict__ lab_s,
                            float* __restrict__ out_labels) {
    if (threadIdx.x != 0 || blockIdx.x != 0) return;
    int P = 0;
    for (int i = 0; i < BB; ++i) P += (labels[i] == 1);
    int p = 0, q = P;
    for (int i = 0; i < BB; ++i) {
        int pos = (labels[i] == 1) ? p++ : q++;
        idx_s[pos]      = nidx[i];
        lab_s[pos]      = labels[i];
        out_labels[pos] = (float)labels[i];
    }
}

// ---------------------------------------------------------------------------
// 2) E [N,D] f32 -> Et [D,N] bf16 (LDS-tiled transpose, coalesced both ways).
// ---------------------------------------------------------------------------
__global__ void transpose_cvt_kernel(const float* __restrict__ E,
                                     bf16* __restrict__ Et) {
    __shared__ float tile[32][33];
    const int n0 = blockIdx.x * 32, d0 = blockIdx.y * 32;
    const int tx = threadIdx.x, ty = threadIdx.y;            // 32 x 8
    #pragma unroll
    for (int i = 0; i < 32; i += 8)
        tile[ty + i][tx] = E[(size_t)(n0 + ty + i) * DD + d0 + tx];
    __syncthreads();
    #pragma unroll
    for (int i = 0; i < 32; i += 8)
        Et[(size_t)(d0 + ty + i) * NN + n0 + tx] = (bf16)tile[tx][ty + i];
}

// ---------------------------------------------------------------------------
// 3) W [L,D,D] f32 -> bf16 (row-major (out,in) already B-fragment friendly).
// ---------------------------------------------------------------------------
__global__ void cvt_w_kernel(const float* __restrict__ W, bf16* __restrict__ Wb, int n) {
    int i = blockIdx.x * blockDim.x + threadIdx.x;
    if (i < n) Wb[i] = (bf16)W[i];
}

// ---------------------------------------------------------------------------
// 4) Inverse row-sums of A, only for positive (sorted) rows.
// ---------------------------------------------------------------------------
__global__ void rowsum_kernel(const float* __restrict__ A,
                              const int* __restrict__ idx_s,
                              const int* __restrict__ lab_s,
                              float* __restrict__ inv_rs) {
    const int r = blockIdx.x;
    if (lab_s[r] == 0) { if (threadIdx.x == 0) inv_rs[r] = 1.0f; return; }
    const int node = idx_s[r];
    const float* row = A + (size_t)node * NN;
    float s = 0.0f;
    for (int j = threadIdx.x; j < NN; j += 256) s += row[j];
    __shared__ float red[256];
    red[threadIdx.x] = s;
    __syncthreads();
    for (int off = 128; off > 0; off >>= 1) {
        if (threadIdx.x < off) red[threadIdx.x] += red[threadIdx.x + off];
        __syncthreads();
    }
    if (threadIdx.x == 0) inv_rs[r] = 1.0f / red[0];
}

// ---------------------------------------------------------------------------
// 5) Aggregation: h0[r,:] = (A[node_r,:]/rowsum) @ E for positives,
//    E[node_r,:] for negatives. One wave = 16 rows x 128 cols. K over N=8192
//    in steps of 32 via v_wmma_f32_16x16x32_bf16; A+B fragments are register
//    double-buffered so each 8-WMMA burst overlaps the next step's A + 8 B-tile loads.
// ---------------------------------------------------------------------------
__global__ __launch_bounds__(128) void agg_kernel(
    const float* __restrict__ A, const float* __restrict__ E,
    const bf16* __restrict__ Et, const int* __restrict__ idx_s,
    const int* __restrict__ lab_s, const float* __restrict__ inv_rs,
    float* __restrict__ h0) {
    const int lane = threadIdx.x & 31;
    const int wid  = blockIdx.x * (blockDim.x >> 5) + (threadIdx.x >> 5);
    const int m0 = (wid >> 1) * 16;      // 256 row-blocks
    const int n0 = (wid & 1) * 128;      // 2 col-groups
    const int lh = lane >> 4;
    const int ml = lane & 15;

    // Sorted labels => block all-negative iff first row negative (uniform).
    if (lab_s[m0] == 0) {
        for (int m = 0; m < 16; ++m) {
            const int node = idx_s[m0 + m];
            const float* src = E + (size_t)node * DD + n0;
            float* dst = h0 + (size_t)(m0 + m) * DD + n0;
            for (int c = lane; c < 128; c += 32) dst[c] = src[c];
        }
        return;
    }
    const bool allpos = (lab_s[m0 + 15] == 1);

    const int nodem = idx_s[m0 + ml];
    const float* arow  = A + (size_t)nodem * NN + 16 * lh;          // kbase for A-frag
    const bf16*  bbase = Et + (size_t)(n0 + ml) * NN + 16 * lh;     // kbase for B-frag

    v8f acc[8] = {};
    v16bf a0, a1, b0[8], b1[8];

    a0 = load_cvt_afrag(arow);
    load_bset(bbase, 0, b0);

    for (int k = 0; k < NN - 64; k += 64) {
        __builtin_prefetch(arow + k + 128, 0, 0);          // global_prefetch_b8
        a1 = load_cvt_afrag(arow + k + 32);
        load_bset(bbase, k + 32, b1);
        #pragma unroll
        for (int t = 0; t < 8; ++t) acc[t] = wmma_bf16(a0, b0[t], acc[t]);
        a0 = load_cvt_afrag(arow + k + 64);
        load_bset(bbase, k + 64, b0);
        #pragma unroll
        for (int t = 0; t < 8; ++t) acc[t] = wmma_bf16(a1, b1[t], acc[t]);
    }
    a1 = load_cvt_afrag(arow + NN - 32);
    load_bset(bbase, NN - 32, b1);
    #pragma unroll
    for (int t = 0; t < 8; ++t) acc[t] = wmma_bf16(a0, b0[t], acc[t]);
    #pragma unroll
    for (int t = 0; t < 8; ++t) acc[t] = wmma_bf16(a1, b1[t], acc[t]);

    float irs[8];
    #pragma unroll
    for (int v = 0; v < 8; ++v) irs[v] = inv_rs[m0 + 8 * lh + v];

    if (allpos) {
        #pragma unroll
        for (int t = 0; t < 8; ++t) {
            const int col = n0 + 16 * t + ml;
            #pragma unroll
            for (int v = 0; v < 8; ++v) {
                const int row = m0 + 8 * lh + v;
                h0[(size_t)row * DD + col] = acc[t][v] * irs[v];
            }
        }
    } else {  // single boundary tile: per-row select
        #pragma unroll
        for (int t = 0; t < 8; ++t) {
            const int col = n0 + 16 * t + ml;
            #pragma unroll
            for (int v = 0; v < 8; ++v) {
                const int row = m0 + 8 * lh + v;
                float val;
                if (lab_s[row] == 1) val = acc[t][v] * irs[v];
                else                 val = E[(size_t)idx_s[row] * DD + col];
                h0[(size_t)row * DD + col] = val;
            }
        }
    }
}

// ---------------------------------------------------------------------------
// 6) MLP layer: hout = relu(hin @ W^T + b). Fully unrolled K=256 (8 steps),
//    batched B-set loads per step so the scheduler pipelines across steps.
// ---------------------------------------------------------------------------
__global__ __launch_bounds__(128) void mlp_kernel(
    const float* __restrict__ hin, const bf16* __restrict__ Wl,
    const float* __restrict__ bias, float* __restrict__ hout) {
    const int lane = threadIdx.x & 31;
    const int wid  = blockIdx.x * (blockDim.x >> 5) + (threadIdx.x >> 5);
    const int m0 = (wid >> 1) * 16;
    const int n0 = (wid & 1) * 128;
    const int lh = lane >> 4;
    const int ml = lane & 15;

    const float* arow  = hin + (size_t)(m0 + ml) * DD + 16 * lh;
    const bf16*  bbase = Wl + (size_t)(n0 + ml) * DD + 16 * lh;

    v8f acc[8] = {};
    #pragma unroll
    for (int k = 0; k < DD; k += 32) {
        v16bf a = load_cvt_afrag(arow + k);
        v16bf bt[8];
        #pragma unroll
        for (int t = 0; t < 8; ++t)
            bt[t] = *(const v16bf*)(bbase + (size_t)t * (16 * DD) + k);
        #pragma unroll
        for (int t = 0; t < 8; ++t) acc[t] = wmma_bf16(a, bt[t], acc[t]);
    }

    #pragma unroll
    for (int t = 0; t < 8; ++t) {
        const int col = n0 + 16 * t + ml;
        const float bv = bias[col];
        #pragma unroll
        for (int v = 0; v < 8; ++v) {
            const int row = m0 + 8 * lh + v;
            float val = acc[t][v] + bv;
            hout[(size_t)row * DD + col] = val > 0.0f ? val : 0.0f;
        }
    }
}

// ---------------------------------------------------------------------------
extern "C" void kernel_launch(void* const* d_in, const int* in_sizes, int n_in,
                              void* d_out, int out_size, void* d_ws, size_t ws_size,
                              hipStream_t stream) {
    (void)in_sizes; (void)n_in; (void)out_size; (void)ws_size;
    const int*   labels = (const int*)d_in[0];
    const int*   nidx   = (const int*)d_in[1];
    const float* A      = (const float*)d_in[2];
    const float* E      = (const float*)d_in[3];
    const float* W      = (const float*)d_in[4];
    const float* bias   = (const float*)d_in[5];

    float* out        = (float*)d_out;
    float* out_labels = out;            // [B]
    float* out_h      = out + BB;       // [B, D]

    char* ws = (char*)d_ws;
    int*   idx_s = (int*)ws;                        ws += (size_t)BB * 4;
    int*   lab_s = (int*)ws;                        ws += (size_t)BB * 4;
    float* inv   = (float*)ws;                      ws += (size_t)BB * 4;
    float* h0    = (float*)ws;                      ws += (size_t)BB * DD * 4;
    float* h1    = (float*)ws;                      ws += (size_t)BB * DD * 4;
    bf16*  Et    = (bf16*)ws;                       ws += (size_t)DD * NN * 2;
    bf16*  Wb    = (bf16*)ws;                       ws += (size_t)LL * DD * DD * 2;

    sort_kernel<<<1, 1, 0, stream>>>(labels, nidx, idx_s, lab_s, out_labels);
    transpose_cvt_kernel<<<dim3(NN / 32, DD / 32), dim3(32, 8), 0, stream>>>(E, Et);
    cvt_w_kernel<<<(LL * DD * DD) / 256, 256, 0, stream>>>(W, Wb, LL * DD * DD);
    rowsum_kernel<<<BB, 256, 0, stream>>>(A, idx_s, lab_s, inv);

    // 512 waves (256 row-blocks x 2 col-groups), 4 waves per block
    agg_kernel<<<(BB / 16) * (DD / 128) / 4, 128, 0, stream>>>(A, E, Et, idx_s, lab_s, inv, h0);

    mlp_kernel<<<128, 128, 0, stream>>>(h0, Wb + 0 * (size_t)DD * DD, bias + 0 * DD, h1);
    mlp_kernel<<<128, 128, 0, stream>>>(h1, Wb + 1 * (size_t)DD * DD, bias + 1 * DD, h0);
    mlp_kernel<<<128, 128, 0, stream>>>(h0, Wb + 2 * (size_t)DD * DD, bias + 2 * DD, out_h);
}
